// SparseMoELayer_13288628814301
// MI455X (gfx1250) — compile-verified, matrix operand order checked by reading, not run on hardware
//
#include <hip/hip_runtime.h>
#include <hip/hip_bf16.h>
#include <cstdint>

#define DIM   1024
#define NEXP  8
#define HDIM  3072
#define SEQ   2048

typedef __attribute__((ext_vector_type(16))) __bf16 v16bf;
typedef __attribute__((ext_vector_type(8)))  float  v8f;
typedef __attribute__((ext_vector_type(4)))  unsigned int   u32x4;
typedef __attribute__((ext_vector_type(4)))  unsigned short u16x4;
typedef __attribute__((ext_vector_type(4)))  float          f32x4;

union V16 { v16bf v; u32x4 q[2]; };

__device__ __forceinline__ unsigned short f2bf(float f) {
    union { float f; uint32_t u; } c; c.f = f;
    uint32_t u = c.u;
    uint32_t r = (u + 0x7FFFu + ((u >> 16) & 1u)) >> 16;   // round-to-nearest-even
    return (unsigned short)r;
}

// ---------------------------------------------------------------------------
// Router: logits = x @ Wr + br ; expert = argmax (first max wins, like jnp);
// build per-expert token lists with atomics.
// ---------------------------------------------------------------------------
__global__ __launch_bounds__(256)
void router_kernel(const float* __restrict__ x, const float* __restrict__ Wr,
                   const float* __restrict__ br, int* __restrict__ counts,
                   int* __restrict__ order) {
    __shared__ float wsh[DIM * NEXP];
    int tid = threadIdx.x;
    for (int i = tid; i < DIM * NEXP; i += 256) wsh[i] = Wr[i];
    __syncthreads();

    int tok = blockIdx.x * 256 + tid;
    const float* xr = x + (size_t)tok * DIM;
    float acc[NEXP];
#pragma unroll
    for (int e = 0; e < NEXP; ++e) acc[e] = br[e];
    for (int d = 0; d < DIM; ++d) {
        float xv = xr[d];
#pragma unroll
        for (int e = 0; e < NEXP; ++e) acc[e] += xv * wsh[d * NEXP + e];
    }
    int best = 0; float bv = acc[0];
#pragma unroll
    for (int e = 1; e < NEXP; ++e) { if (acc[e] > bv) { bv = acc[e]; best = e; } }
    int slot = atomicAdd(&counts[best], 1);
    order[best * SEQ + slot] = tok;
}

// ---------------------------------------------------------------------------
// x fp32 -> bf16 (row-major, unchanged layout)
// ---------------------------------------------------------------------------
__global__ __launch_bounds__(256)
void cvt_x_kernel(const float* __restrict__ x, unsigned short* __restrict__ xb) {
    int i = (blockIdx.x * 256 + threadIdx.x) * 4;
    f32x4 v = *(const f32x4*)(x + i);
    u16x4 o;
    o.x = f2bf(v.x); o.y = f2bf(v.y); o.z = f2bf(v.z); o.w = f2bf(v.w);
    *(u16x4*)(xb + i) = o;
}

// ---------------------------------------------------------------------------
// Per-expert transpose+convert: in fp32 [R][C]  ->  out bf16 [C][R]
// (weights become [N][K] so the WMMA B operand is a pair of contiguous B128s)
// ---------------------------------------------------------------------------
__global__ __launch_bounds__(256)
void transpose_bf16_kernel(const float* __restrict__ in, unsigned short* __restrict__ out,
                           int R, int C) {
    __shared__ unsigned short tile[32][33];
    const float*    src = in  + (size_t)blockIdx.z * R * C;
    unsigned short* dst = out + (size_t)blockIdx.z * R * C;
    int c0 = blockIdx.x * 32, r0 = blockIdx.y * 32;
    int tx = threadIdx.x, ty = threadIdx.y;         // block = (32, 8)
#pragma unroll
    for (int j = 0; j < 32; j += 8)
        tile[ty + j][tx] = f2bf(src[(size_t)(r0 + ty + j) * C + c0 + tx]);
    __syncthreads();
#pragma unroll
    for (int j = 0; j < 32; j += 8)
        dst[(size_t)(c0 + ty + j) * R + r0 + tx] = tile[tx][ty + j];
}

// ---------------------------------------------------------------------------
// Tiled expert GEMM using V_WMMA_F32_16X16X32_BF16.
//   A: bf16 rows indexed by token id, stride K   (x or h buffer)
//   Wt: bf16 [E][N][K]  (pre-transposed weights)
//   out row per token, stride N. GELU => bf16 out (h), else fp32 out (y).
// Block: 256 threads (8 waves). Tile: M=64 tokens x N=128 cols.
// Wave w owns N-subtile w; 4 M-subtile accumulators (4 x v8f).
// X tile staged into LDS with GLOBAL_LOAD_ASYNC_TO_LDS_B128 (ASYNCcnt).
// ---------------------------------------------------------------------------
template <bool GELU, typename OUT>
__global__ __launch_bounds__(256)
void moe_gemm_kernel(const unsigned short* __restrict__ A,
                     const unsigned short* __restrict__ Wt,
                     const float* __restrict__ bias,
                     const int* __restrict__ order,
                     const int* __restrict__ counts,
                     OUT* __restrict__ out, int K, int N) {
    int e   = blockIdx.z;
    int cnt = counts[e];
    int mt  = blockIdx.y;
    if (mt * 64 >= cnt) return;
    int tileCnt = cnt - mt * 64; if (tileCnt > 64) tileCnt = 64;

    __shared__ int toks[64];
    __shared__ __align__(16) unsigned short xs[64][264];  // 256-wide K chunk, padded stride

    int tid = threadIdx.x;
    if (tid < 64) {
        int s = mt * 64 + tid;
        toks[tid] = order[e * SEQ + ((tid < tileCnt) ? s : mt * 64)];  // dup first token as dummy
    }
    __syncthreads();

    int lane = tid & 31;
    int wv   = tid >> 5;
    int lhi  = lane >> 4;        // 0 / 1  -> K half selector
    int lrow = lane & 15;        // M (A) or N (B) within subtile
    int nG   = blockIdx.x * 128 + wv * 16;
    const unsigned short* Brow = Wt + ((size_t)e * N + nG + lrow) * K + lhi * 8;

    v8f acc[4] = {};

    for (int kc = 0; kc < K; kc += 256) {
        // speculative prefetch of this wave's next B K-chunk (global_prefetch_b8)
        {
            int kp = (kc + 256 < K) ? (kc + 256) : kc;
            __builtin_prefetch(Brow + kp, 0, 1);
        }
        // async-stage X chunk [64][256] into LDS: each lane copies 8 x 16B
        // directly memory->LDS (no VGPR round trip), tracked by ASYNCcnt.
#pragma unroll
        for (int i = 0; i < 8; ++i) {
            int q   = tid + i * 256;     // 0..2047 quad-transfers
            int row = q >> 5;
            int qc  = q & 31;
            const unsigned short* src = A + (size_t)toks[row] * K + kc + qc * 8;
            unsigned ldsa = (unsigned)(size_t)&xs[row][qc * 8];   // wave-relative LDS offset
            asm volatile("global_load_async_to_lds_b128 %0, %1, off"
                         :: "v"(ldsa), "v"(src) : "memory");
        }
        asm volatile("s_wait_asynccnt 0" ::: "memory");
        __syncthreads();

#pragma unroll
        for (int kk = 0; kk < 256; kk += 32) {
            V16 b;
            b.q[0] = *(const u32x4*)(Brow + kc + kk);
            b.q[1] = *(const u32x4*)(Brow + kc + kk + 16);
            V16 a[4];
#pragma unroll
            for (int m = 0; m < 4; ++m) {
                const unsigned short* ap = &xs[m * 16 + lrow][kk + lhi * 8];
                a[m].q[0] = *(const u32x4*)ap;
                a[m].q[1] = *(const u32x4*)(ap + 16);
            }
#pragma unroll
            for (int m = 0; m < 4; ++m)
                acc[m] = __builtin_amdgcn_wmma_f32_16x16x32_bf16(
                    false, a[m].v, false, b.v, (short)0, acc[m], false, false);
        }
        __syncthreads();
    }

    // epilogue: D layout -> lane col = nG + lane%16 ; row M = r + 8*(lane>=16)
    int col  = nG + lrow;
    float bv = bias[(size_t)e * N + col];
#pragma unroll
    for (int m = 0; m < 4; ++m) {
#pragma unroll
        for (int r = 0; r < 8; ++r) {
            int slot = m * 16 + r + lhi * 8;
            if (slot < tileCnt) {
                float v = acc[m][r] + bv;
                if (GELU) v = 0.5f * v * (1.0f + erff(v * 0.70710678118654752f));
                size_t off = (size_t)toks[slot] * N + col;
                if constexpr (sizeof(OUT) == 2) out[off] = (OUT)f2bf(v);
                else                            out[off] = (OUT)v;
            }
        }
    }
}

// ---------------------------------------------------------------------------
extern "C" void kernel_launch(void* const* d_in, const int* in_sizes, int n_in,
                              void* d_out, int out_size, void* d_ws, size_t ws_size,
                              hipStream_t stream) {
    const float* x  = (const float*)d_in[0];  // [1,2048,1024]
    const float* Wr = (const float*)d_in[1];  // [1024,8]
    const float* br = (const float*)d_in[2];  // [8]
    const float* W1 = (const float*)d_in[3];  // [8,1024,3072]
    const float* b1 = (const float*)d_in[4];  // [8,3072]
    const float* W2 = (const float*)d_in[5];  // [8,3072,1024]
    const float* b2 = (const float*)d_in[6];  // [8,1024]
    float* y = (float*)d_out;                 // [1,2048,1024]

    // workspace carving (256B aligned)
    uint8_t* ws = (uint8_t*)d_ws;
    size_t off = 0;
    auto carve = [&](size_t bytes) -> void* {
        void* p = ws + off;
        off = (off + bytes + 255) & ~(size_t)255;
        return p;
    };
    int*            counts = (int*)carve(NEXP * sizeof(int));
    int*            order  = (int*)carve((size_t)NEXP * SEQ * sizeof(int));
    unsigned short* xb     = (unsigned short*)carve((size_t)SEQ * DIM * 2);
    unsigned short* hbuf   = (unsigned short*)carve((size_t)SEQ * HDIM * 2);
    unsigned short* W1t    = (unsigned short*)carve((size_t)NEXP * DIM * HDIM * 2);  // [E][3072][1024]
    unsigned short* W2t    = (unsigned short*)carve((size_t)NEXP * DIM * HDIM * 2);  // [E][1024][3072]

    hipMemsetAsync(counts, 0, NEXP * sizeof(int), stream);

    router_kernel<<<SEQ / 256, 256, 0, stream>>>(x, Wr, br, counts, order);

    cvt_x_kernel<<<(SEQ * DIM) / (256 * 4), 256, 0, stream>>>(x, xb);

    // W1 [e][1024][3072] -> W1t [e][3072][1024]
    transpose_bf16_kernel<<<dim3(HDIM / 32, DIM / 32, NEXP), dim3(32, 8), 0, stream>>>(
        W1, W1t, DIM, HDIM);
    // W2 [e][3072][1024] -> W2t [e][1024][3072]
    transpose_bf16_kernel<<<dim3(DIM / 32, HDIM / 32, NEXP), dim3(32, 8), 0, stream>>>(
        W2, W2t, HDIM, DIM);

    // h = gelu(x @ W1 + b1)  -> bf16 hbuf rows by token id
    moe_gemm_kernel<true, unsigned short>
        <<<dim3(HDIM / 128, SEQ / 64, NEXP), 256, 0, stream>>>(
            xb, W1t, b1, order, counts, hbuf, DIM, HDIM);

    // y = h @ W2 + b2 -> fp32 d_out rows by token id
    moe_gemm_kernel<false, float>
        <<<dim3(DIM / 128, SEQ / 64, NEXP), 256, 0, stream>>>(
            hbuf, W2t, b2, order, counts, y, HDIM, DIM);
}
